// MaskedLinear_24592982736969
// MI455X (gfx1250) — matrix operand order repaired and checked by reading, not compiled, not run on hardware
//
#include <hip/hip_runtime.h>
#include <hip/hip_bf16.h>

typedef __attribute__((ext_vector_type(16))) __bf16 bf16x16;
typedef __attribute__((ext_vector_type(8)))  __bf16 bf16x8;
typedef __attribute__((ext_vector_type(8)))  float  f32x8;

#define N_DIM 1024

__device__ __forceinline__ __bf16 f2bf(float f) { return (__bf16)f; }

// ---------------------------------------------------------------------------
// Pass 1: w_bf16[m,k] = bf16(weight[m,k] * mask[m,k]) ; 8 elements per thread
// ---------------------------------------------------------------------------
__global__ __launch_bounds__(256)
void MaskedLinear_wprep(const float* __restrict__ weight,
                        const float* __restrict__ mask,
                        __bf16* __restrict__ wbf)
{
    const int t = (blockIdx.x * blockDim.x + threadIdx.x) * 8;
    float4 w0 = *(const float4*)(weight + t);
    float4 w1 = *(const float4*)(weight + t + 4);
    float4 m0 = *(const float4*)(mask + t);
    float4 m1 = *(const float4*)(mask + t + 4);
    bf16x8 r;
    r[0] = f2bf(w0.x * m0.x);
    r[1] = f2bf(w0.y * m0.y);
    r[2] = f2bf(w0.z * m0.z);
    r[3] = f2bf(w0.w * m0.w);
    r[4] = f2bf(w1.x * m1.x);
    r[5] = f2bf(w1.y * m1.y);
    r[6] = f2bf(w1.z * m1.z);
    r[7] = f2bf(w1.w * m1.w);
    *(bf16x8*)(wbf + t) = r;
}

// ---------------------------------------------------------------------------
// Pass 2: out = x @ w_bf16^T + bias using v_wmma_f32_16x16x32_bf16
// Block: 256 threads = 8 waves arranged 4(M) x 2(N); block tile 256x128.
// Wave tile 64x64 = 4x4 WMMA accumulators.
// ---------------------------------------------------------------------------
__global__ __launch_bounds__(256, 1)
void MaskedLinear_wmma_gemm(const float* __restrict__ x,
                            const __bf16* __restrict__ w,
                            const float* __restrict__ bias,
                            float* __restrict__ out)
{
    const int tid  = threadIdx.x;
    const int lane = tid & 31;
    const int wid  = tid >> 5;
    const int lrow = lane & 15;   // row (A) / col (B) within tile
    const int lsel = lane >> 4;   // which K-half this lane holds

    const int wave_m = wid & 3;   // 0..3
    const int wave_n = wid >> 2;  // 0..1

    const int row_base = blockIdx.y * 256 + wave_m * 64;  // batch rows
    const int col_base = blockIdx.x * 128 + wave_n * 64;  // output features

    f32x8 acc[4][4] = {};

    const float*  xrow[4];
    const __bf16* wrow[4];
#pragma unroll
    for (int i = 0; i < 4; ++i)
        xrow[i] = x + (size_t)(row_base + i * 16 + lrow) * N_DIM;
#pragma unroll
    for (int j = 0; j < 4; ++j)
        wrow[j] = w + (size_t)(col_base + j * 16 + lrow) * N_DIM;

    for (int k0 = 0; k0 < N_DIM; k0 += 32) {
        // ---- A fragments: x rows, fp32 -> bf16 in-register ----
        // A layout: elems 0-7 -> K = k0 + lsel*8 + e ; elems 8-15 -> +16
        const int ka0 = k0 + lsel * 8;
        const int ka1 = ka0 + 16;
        bf16x16 afrag[4];
#pragma unroll
        for (int i = 0; i < 4; ++i) {
            float4 c0 = *(const float4*)(xrow[i] + ka0);
            float4 c1 = *(const float4*)(xrow[i] + ka0 + 4);
            float4 c2 = *(const float4*)(xrow[i] + ka1);
            float4 c3 = *(const float4*)(xrow[i] + ka1 + 4);
            bf16x16 a;
            a[0]  = f2bf(c0.x); a[1]  = f2bf(c0.y); a[2]  = f2bf(c0.z); a[3]  = f2bf(c0.w);
            a[4]  = f2bf(c1.x); a[5]  = f2bf(c1.y); a[6]  = f2bf(c1.z); a[7]  = f2bf(c1.w);
            a[8]  = f2bf(c2.x); a[9]  = f2bf(c2.y); a[10] = f2bf(c2.z); a[11] = f2bf(c2.w);
            a[12] = f2bf(c3.x); a[13] = f2bf(c3.y); a[14] = f2bf(c3.z); a[15] = f2bf(c3.w);
            afrag[i] = a;
        }

        // ---- B fragments: w rows (already bf16) ----
        // B layout: elems 0-7 -> K = k0 + lsel*8 + e ; elems 8-15 -> +16
        const int kb = k0 + lsel * 8;
        bf16x16 bfrag[4];
#pragma unroll
        for (int j = 0; j < 4; ++j) {
            bf16x8 b0 = *(const bf16x8*)(wrow[j] + kb);
            bf16x8 b1 = *(const bf16x8*)(wrow[j] + kb + 16);
            bf16x16 bb;
#pragma unroll
            for (int e = 0; e < 8; ++e) { bb[e] = b0[e]; bb[8 + e] = b1[e]; }
            bfrag[j] = bb;
        }

        // ---- 16 WMMAs ----
#pragma unroll
        for (int i = 0; i < 4; ++i)
#pragma unroll
            for (int j = 0; j < 4; ++j)
                acc[i][j] = __builtin_amdgcn_wmma_f32_16x16x32_bf16(
                    /*neg_a=*/false, afrag[i],
                    /*neg_b=*/false, bfrag[j],
                    /*c_mod=*/(short)0, acc[i][j],
                    /*reuse_a=*/false, /*reuse_b=*/false);
    }

    // ---- epilogue: bias add + store ----
    // D layout: VGPR v, lane l -> row = v + 8*(l>>4), col = l&15
#pragma unroll
    for (int j = 0; j < 4; ++j) {
        const int col = col_base + j * 16 + lrow;
        const float bv = bias[col];
#pragma unroll
        for (int i = 0; i < 4; ++i) {
#pragma unroll
            for (int v = 0; v < 8; ++v) {
                const int row = row_base + i * 16 + v + lsel * 8;
                out[(size_t)row * N_DIM + col] = acc[i][j][v] + bv;
            }
        }
    }
}

// ---------------------------------------------------------------------------
extern "C" void kernel_launch(void* const* d_in, const int* in_sizes, int n_in,
                              void* d_out, int out_size, void* d_ws, size_t ws_size,
                              hipStream_t stream)
{
    const float* x      = (const float*)d_in[0];   // [B, N] fp32
    const float* mask   = (const float*)d_in[1];   // [N, N] fp32
    const float* weight = (const float*)d_in[2];   // [N, N] fp32
    const float* bias   = (const float*)d_in[3];   // [N]    fp32
    float* out = (float*)d_out;                    // [B, N] fp32
    __bf16* wbf = (__bf16*)d_ws;                   // [N, N] bf16 scratch (2 MB)

    const int nw = in_sizes[1];                    // N*N = 1048576
    const int rows = in_sizes[0] / N_DIM;          // B = 65536

    // Pass 1: masked weight -> bf16 (8 elems/thread)
    MaskedLinear_wprep<<<nw / (256 * 8), 256, 0, stream>>>(weight, mask, wbf);

    // Pass 2: WMMA GEMM. grid.x covers N/128 feature tiles, grid.y covers B/256 rows.
    dim3 grid(N_DIM / 128, rows / 256);
    MaskedLinear_wmma_gemm<<<grid, 256, 0, stream>>>(x, wbf, bias, out);
}
